// LSTMAttention_7687991459888
// MI455X (gfx1250) — matrix operand order repaired and checked by reading, not compiled, not run on hardware
//
#include <hip/hip_runtime.h>

#define B_ 64
#define T_ 256
#define S_ 256
#define H_ 1024
#define I_ 1024
#define G4 4096  /* 4*H */

typedef __attribute__((ext_vector_type(4)))  __bf16 v4bf;
typedef __attribute__((ext_vector_type(8)))  __bf16 v8bf;
typedef __attribute__((ext_vector_type(16))) __bf16 v16bf;
typedef __attribute__((ext_vector_type(4)))  float  v4f;
typedef __attribute__((ext_vector_type(8)))  float  v8f;

// gfx1250 has V_TANH_F32 (single TRANS op) — use it when the builtin exists.
__device__ __forceinline__ float fast_tanh(float x) {
#if __has_builtin(__builtin_amdgcn_tanhf)
  return __builtin_amdgcn_tanhf(x);
#else
  return tanhf(x);
#endif
}
// sigmoid(x) = 0.5*tanh(x/2) + 0.5  -> one TRANS op
__device__ __forceinline__ float fast_sigmoid(float x) {
  return __builtin_fmaf(0.5f, fast_tanh(0.5f * x), 0.5f);
}

// ---------------- conversion / copy (x4 vectorized) ----------------
__global__ __launch_bounds__(256)
void k_f32_to_bf16(const float* __restrict__ in, __bf16* __restrict__ out, long n4)
{
  long i = (long)blockIdx.x * blockDim.x + threadIdx.x;
  if (i < n4) {
    v4f f = *(const v4f*)(in + i * 4);
    v4bf o;
    #pragma unroll
    for (int j = 0; j < 4; ++j) o[j] = (__bf16)f[j];
    *(v4bf*)(out + i * 4) = o;
  }
}

__global__ __launch_bounds__(256)
void k_copy_f32(const float* __restrict__ in, float* __restrict__ out, long n4)
{
  long i = (long)blockIdx.x * blockDim.x + threadIdx.x;
  if (i < n4) *(v4f*)(out + i * 4) = *(const v4f*)(in + i * 4);
}

// ---------------- WMMA GEMM ----------------
// C[M,N] = A[M,K] @ W[N,K]^T  (+bias1[n]) (+bias2[n]) (+add[m,n]) (optional tanh)
// Block = 256 threads = 8 waves, arranged 4 (M) x 2 (N).
// Wave tile = 16 x (NT*16).  Block tile = 64 x (NT*32).
template<int NT, bool TANH, bool OUTBF16>
__global__ __launch_bounds__(256)
void k_gemm_wmma(const __bf16* __restrict__ A, long lda,
                 const __bf16* __restrict__ W, long ldw,
                 float* __restrict__ C, __bf16* __restrict__ Cb, long ldc,
                 const float* __restrict__ bias1, const float* __restrict__ bias2,
                 const float* __restrict__ add, long ld_add, int K)
{
  const int lane = threadIdx.x & 31;
  const int wave = threadIdx.x >> 5;
  const int half = lane >> 4;    // which 16-lane half of the wave
  const int l15  = lane & 15;
  const int m0 = blockIdx.x * 64 + (wave >> 1) * 16;
  const int n0 = blockIdx.y * (NT * 32) + (wave & 1) * (NT * 16);

  v8f acc[NT] = {};

  // A fragment (16x32 bf16): lane<16 -> K {0..7,16..23}; lane>=16 -> K {8..15,24..31}
  const __bf16* ap  = A + (size_t)(m0 + l15) * lda + half * 8;
  // B fragment (32x16 bf16) from row-major W[N,K]: lane<16 -> K {0..7,16..23}; lane>=16 -> K {8..15,24..31}
  const __bf16* bp0 = W + (size_t)(n0 + l15) * ldw + half * 8;

  for (int k = 0; k < K; k += 32) {
    v16bf Af;
    {
      v8bf a0 = *(const v8bf*)(ap + k);
      v8bf a1 = *(const v8bf*)(ap + k + 16);
      #pragma unroll
      for (int i = 0; i < 8; ++i) { Af[i] = a0[i]; Af[i + 8] = a1[i]; }
    }
    #pragma unroll
    for (int j = 0; j < NT; ++j) {
      const __bf16* bp = bp0 + (size_t)(16 * j) * ldw + k;
      v8bf b0 = *(const v8bf*)(bp);
      v8bf b1 = *(const v8bf*)(bp + 16);
      v16bf Bf;
      #pragma unroll
      for (int i = 0; i < 8; ++i) { Bf[i] = b0[i]; Bf[i + 8] = b1[i]; }
      acc[j] = __builtin_amdgcn_wmma_f32_16x16x32_bf16(
          false, Af, false, Bf, (short)0, acc[j], false, false);
    }
  }

  // C/D layout: VGPR r, lane -> (m = m0 + half*8 + r, n = n0 + 16j + l15)
  #pragma unroll
  for (int j = 0; j < NT; ++j) {
    const int n = n0 + 16 * j + l15;
    float bia = 0.f;
    if (bias1) bia += bias1[n];
    if (bias2) bia += bias2[n];
    #pragma unroll
    for (int r = 0; r < 8; ++r) {
      const int m = m0 + half * 8 + r;
      float v = acc[j][r] + bia;
      if (add)  v += add[(size_t)m * ld_add + n];
      if (TANH) v = fast_tanh(v);
      if (OUTBF16) Cb[(size_t)m * ldc + n] = (__bf16)v;
      else         C [(size_t)m * ldc + n] = v;
    }
  }
}

// ---------------- LSTM cell (elementwise over B*H) ----------------
__global__ __launch_bounds__(256)
void k_lstm_cell(const float* __restrict__ gates,   // [B,4H] (i|f|g|o)
                 float* __restrict__ c,             // in: c_{t-1}, out: c_t  [B*H]
                 float* __restrict__ hout, long ld_hout, // h_out[b, t, :] slice
                 __bf16* __restrict__ hxb,          // bf16 hy for next-step GEMMs
                 __bf16* __restrict__ catb)         // [B,2H]: hy into cols 1024..2047
{
  int i = blockIdx.x * blockDim.x + threadIdx.x;    // 0 .. B*H-1
  int b = i >> 10, h = i & 1023;
  const float* g = gates + (size_t)b * G4;
  float ig = fast_sigmoid(g[h]);
  float fg = fast_sigmoid(g[h + 1024]);
  float gg = fast_tanh(g[h + 2048]);
  float og = fast_sigmoid(g[h + 3072]);
  float cy = fg * c[i] + ig * gg;
  float hy = og * fast_tanh(cy);
  c[i] = cy;
  hout[(size_t)b * ld_hout + h] = hy;
  hxb[i] = (__bf16)hy;
  catb[(size_t)b * (2 * H_) + H_ + h] = (__bf16)hy;
}

// ---------------- attention scores: e[b,s] = sum_h tanh(target+source)*Wa_v ----------------
// One wave per (b,s); lane owns 8 consecutive h per iteration (v8bf = 512B/wave/load).
__global__ __launch_bounds__(256)
void k_attn_scores(const float* __restrict__ target,   // [B,H]
                   const __bf16* __restrict__ srcb,    // [B,S,H] bf16 (L2-resident)
                   const float* __restrict__ wav,      // [H]
                   float* __restrict__ e)              // [B,S]
{
  int wave = threadIdx.x >> 5, lane = threadIdx.x & 31;
  int p = blockIdx.x * 8 + wave;      // flat (b*S + s)
  int b = p >> 8;
  const __bf16* sp = srcb + (size_t)p * H_ + lane * 8;
  const float*  tp = target + (size_t)b * H_ + lane * 8;
  const float*  wp = wav + lane * 8;
  float acc = 0.f;
  #pragma unroll
  for (int it = 0; it < H_ / 256; ++it) {   // 4 iterations
    const int h = it * 256;
    v8bf s8 = *(const v8bf*)(sp + h);
    v4f  t0 = *(const v4f*)(tp + h);
    v4f  t1 = *(const v4f*)(tp + h + 4);
    v4f  w0 = *(const v4f*)(wp + h);
    v4f  w1 = *(const v4f*)(wp + h + 4);
    #pragma unroll
    for (int j = 0; j < 4; ++j) acc += fast_tanh(t0[j] + (float)s8[j])     * w0[j];
    #pragma unroll
    for (int j = 0; j < 4; ++j) acc += fast_tanh(t1[j] + (float)s8[j + 4]) * w1[j];
  }
  #pragma unroll
  for (int off = 16; off; off >>= 1) acc += __shfl_xor(acc, off, 32);
  if (lane == 0) e[p] = acc;
}

// ---------------- softmax over S (one block per b; S==blockDim==256) ----------------
__global__ __launch_bounds__(256)
void k_softmax(const float* __restrict__ e, float* __restrict__ attn,
               float* __restrict__ aout, long ld_ao)
{
  __shared__ float red[8];
  const int b = blockIdx.x, s = threadIdx.x;
  const int lane = s & 31, w = s >> 5;
  float v = e[b * S_ + s];
  float m = v;
  #pragma unroll
  for (int off = 16; off; off >>= 1) m = fmaxf(m, __shfl_xor(m, off, 32));
  if (lane == 0) red[w] = m;
  __syncthreads();
  float bm = red[0];
  #pragma unroll
  for (int i = 1; i < 8; ++i) bm = fmaxf(bm, red[i]);
  float ex = __expf(v - bm);
  __syncthreads();
  float sum = ex;
  #pragma unroll
  for (int off = 16; off; off >>= 1) sum += __shfl_xor(sum, off, 32);
  if (lane == 0) red[w] = sum;
  __syncthreads();
  float bs = 0.f;
  #pragma unroll
  for (int i = 0; i < 8; ++i) bs += red[i];
  float a = ex / bs;
  attn[b * S_ + s] = a;
  aout[(size_t)b * ld_ao + s] = a;
}

// ---------------- context: c[b,h] = sum_s attn[b,s]*ctx[b,s,h] ----------------
// One block per b; each thread owns 4 h (v4bf loads, fully coalesced).
// attn row staged global->LDS with the gfx1250 async-to-LDS path (ASYNCcnt).
__global__ __launch_bounds__(256)
void k_context(const float* __restrict__ attn, const __bf16* __restrict__ ctxb,
               float* __restrict__ cout, long ld_co, __bf16* __restrict__ catb)
{
  __shared__ float a[S_];
  const int b = blockIdx.x;
  const int h0 = threadIdx.x * 4;
  {
    // global_load_async_to_lds_b32: VDST = LDS byte address, VADDR = 64-bit global addr
    unsigned lds_off = (unsigned)(size_t)(&a[threadIdx.x]);
    unsigned long long gaddr =
        (unsigned long long)(const void*)(attn + (size_t)b * S_ + threadIdx.x);
    asm volatile("global_load_async_to_lds_b32 %0, %1, off"
                 :: "v"(lds_off), "v"(gaddr) : "memory");
    asm volatile("s_wait_asynccnt 0" ::: "memory");
  }
  __syncthreads();

  v4f acc = {0.f, 0.f, 0.f, 0.f};
  const __bf16* cp = ctxb + (size_t)b * S_ * H_ + h0;
  #pragma unroll 4
  for (int s = 0; s < S_; ++s) {
    float as = a[s];
    v4bf c4 = *(const v4bf*)(cp + (size_t)s * H_);
    #pragma unroll
    for (int j = 0; j < 4; ++j) acc[j] += as * (float)c4[j];
  }
  *(v4f*)(cout + (size_t)b * ld_co + h0) = acc;
  v4bf cb;
  #pragma unroll
  for (int j = 0; j < 4; ++j) cb[j] = (__bf16)acc[j];
  *(v4bf*)(catb + (size_t)b * (2 * H_) + h0) = cb;
}

// ---------------- final hy/cy outputs ----------------
__global__ __launch_bounds__(256)
void k_finalize(const float* __restrict__ hout, const float* __restrict__ c,
                float* __restrict__ hy, float* __restrict__ cy)
{
  int i = blockIdx.x * blockDim.x + threadIdx.x;   // 0 .. B*H-1
  int b = i >> 10, h = i & 1023;
  hy[i] = hout[((size_t)b * T_ + (T_ - 1)) * H_ + h];
  cy[i] = c[i];
}

// =====================================================================
extern "C" void kernel_launch(void* const* d_in, const int* in_sizes, int n_in,
                              void* d_out, int out_size, void* d_ws, size_t ws_size,
                              hipStream_t stream)
{
  (void)in_sizes; (void)n_in; (void)out_size; (void)ws_size;

  const float* input = (const float*)d_in[0];
  const float* hx0   = (const float*)d_in[1];
  const float* cx0   = (const float*)d_in[2];
  const float* ctx   = (const float*)d_in[3];
  const float* W_in  = (const float*)d_in[4];
  const float* b_in  = (const float*)d_in[5];
  const float* W_hid = (const float*)d_in[6];
  const float* b_hid = (const float*)d_in[7];
  const float* Wa_in = (const float*)d_in[8];
  const float* Wa_c  = (const float*)d_in[9];
  const float* ba_c  = (const float*)d_in[10];
  const float* Wa_v  = (const float*)d_in[11];
  const float* Wa_out= (const float*)d_in[12];

  // ----- output slices (flat, reference return order) -----
  float* out  = (float*)d_out;
  const size_t BTH = (size_t)B_ * T_ * H_;
  const size_t BTS = (size_t)B_ * T_ * S_;
  float* d_h  = out;
  float* d_ht = out + BTH;
  float* d_c  = out + 2 * BTH;
  float* d_at = out + 3 * BTH;
  float* d_hy = d_at + BTS;
  float* d_cy = d_hy + (size_t)B_ * H_;

  // ----- workspace carve -----
  char* p = (char*)d_ws;
  auto carve = [&](size_t bytes) -> char* {
    char* r = p; p += (bytes + 255) & ~(size_t)255; return r;
  };
  const size_t nIn   = (size_t)B_ * T_ * I_;
  const size_t nCtx  = (size_t)B_ * S_ * H_;
  __bf16* inb    = (__bf16*)carve(nIn * 2);
  __bf16* ctxb   = (__bf16*)carve(nCtx * 2);
  __bf16* winb   = (__bf16*)carve((size_t)G4 * I_ * 2);
  __bf16* whidb  = (__bf16*)carve((size_t)G4 * H_ * 2);
  __bf16* wainb  = (__bf16*)carve((size_t)H_ * H_ * 2);
  __bf16* wacb   = (__bf16*)carve((size_t)H_ * H_ * 2);
  __bf16* waoutb = (__bf16*)carve((size_t)H_ * 2 * H_ * 2);
  __bf16* srcb   = (__bf16*)carve(nCtx * 2);                 // source, bf16, L2-resident
  float*  gatesx = (float*) carve((size_t)B_ * T_ * G4 * 4); // x@W_in^T + biases, all t
  __bf16* hxb    = (__bf16*)carve((size_t)B_ * H_ * 2);
  float*  cx     = (float*) carve((size_t)B_ * H_ * 4);
  float*  gates  = (float*) carve((size_t)B_ * G4 * 4);
  float*  target = (float*) carve((size_t)B_ * H_ * 4);
  float*  e      = (float*) carve((size_t)B_ * S_ * 4);
  float*  attn   = (float*) carve((size_t)B_ * S_ * 4);
  __bf16* catb   = (__bf16*)carve((size_t)B_ * 2 * H_ * 2);

  const dim3 blk(256);
  auto cvt = [&](const float* src, __bf16* dst, long n) {
    long n4 = n / 4;
    k_f32_to_bf16<<<dim3((unsigned)((n4 + 255) / 256)), blk, 0, stream>>>(src, dst, n4);
  };

  // ----- precompute (no sequential dependence) -----
  cvt(input, inb,    (long)nIn);
  cvt(ctx,   ctxb,   (long)nCtx);
  cvt(W_in,  winb,   (long)G4 * I_);
  cvt(W_hid, whidb,  (long)G4 * H_);
  cvt(Wa_in, wainb,  (long)H_ * H_);
  cvt(Wa_c,  wacb,   (long)H_ * H_);
  cvt(Wa_out,waoutb, (long)H_ * 2 * H_);
  cvt(hx0,   hxb,    (long)B_ * H_);
  k_copy_f32<<<dim3((B_ * H_ / 4) / 256), blk, 0, stream>>>(cx0, cx, (long)B_ * H_ / 4);

  // gatesx[b*T+t, :] = x @ W_in^T + (b_in + b_hid)   [16384 x 4096], K=1024
  k_gemm_wmma<4, false, false><<<dim3((B_ * T_) / 64, G4 / 128), blk, 0, stream>>>(
      inb, I_, winb, I_, gatesx, nullptr, G4, b_in, b_hid, nullptr, 0, I_);
  // source = ctx @ Wa_c^T + ba_c  ->  bf16  [16384 x 1024], K=1024
  k_gemm_wmma<4, false, true><<<dim3((B_ * S_) / 64, H_ / 128), blk, 0, stream>>>(
      ctxb, H_, wacb, H_, nullptr, srcb, H_, ba_c, nullptr, nullptr, 0, H_);

  // ----- sequential recurrence -----
  for (int t = 0; t < T_; ++t) {
    // gates = hx @ W_hid^T + gatesx[:, t, :]   [64 x 4096], K=1024  (128 blocks)
    k_gemm_wmma<1, false, false><<<dim3(1, G4 / 32), blk, 0, stream>>>(
        hxb, H_, whidb, H_, gates, nullptr, G4, nullptr, nullptr,
        gatesx + (size_t)t * G4, (long)T_ * G4, H_);
    // LSTM cell -> cy, hy (writes h_out[:,t,:], hxb, catb[:,H:2H])
    k_lstm_cell<<<dim3((B_ * H_) / 256), blk, 0, stream>>>(
        gates, cx, d_h + (size_t)t * H_, (long)T_ * H_, hxb, catb);
    // target = hy @ Wa_in^T   [64 x 1024], K=1024
    k_gemm_wmma<1, false, false><<<dim3(1, H_ / 32), blk, 0, stream>>>(
        hxb, H_, wainb, H_, target, nullptr, H_, nullptr, nullptr, nullptr, 0, H_);
    // e[b,s] = sum_h tanh(target + source) * Wa_v
    k_attn_scores<<<dim3((B_ * S_) / 8), blk, 0, stream>>>(target, srcb, Wa_v, e);
    // softmax over S; writes attn and attn_out[:,t,:]
    k_softmax<<<dim3(B_), blk, 0, stream>>>(e, attn, d_at + (size_t)t * S_, (long)T_ * S_);
    // c = attn @ ctx; writes c_out[:,t,:] and catb[:,0:H]
    k_context<<<dim3(B_), blk, 0, stream>>>(
        attn, ctxb, d_c + (size_t)t * H_, (long)T_ * H_, catb);
    // h_tilde = tanh([c,hy] @ Wa_out^T) -> h_tilde_out[:,t,:]   [64 x 1024], K=2048
    k_gemm_wmma<1, true, false><<<dim3(1, H_ / 32), blk, 0, stream>>>(
        catb, 2 * H_, waoutb, 2 * H_, d_ht + (size_t)t * H_, nullptr, (long)T_ * H_,
        nullptr, nullptr, nullptr, 0, 2 * H_);
  }

  // hy = h_out[:, T-1, :], cy = c buffer
  k_finalize<<<dim3((B_ * H_) / 256), blk, 0, stream>>>(d_h, cx, d_hy, d_cy);
}
